// BipartiteSAGE2mod_78159814852630
// MI455X (gfx1250) — matrix operand order repaired and checked by reading, not compiled, not run on hardware
//
#include <hip/hip_runtime.h>
#include <hip/hip_bf16.h>

typedef __attribute__((ext_vector_type(16))) _Float16 v16h;
typedef __attribute__((ext_vector_type(8)))  float    v8f;

// ---------------------------------------------------------------------------
// Fused WMMA GEMM:  C[M, NT*16] = op( A[M,K] @ B[K, NT*16] + bias + Cin )
//   K, LDB (B row stride), NT (column tiles), RS (fused row-scale) are
//   compile-time constants.
//   - B is staged once per block into LDS, pre-swizzled into the exact
//     v_wmma_f32_16x16x32_f16 B-fragment layout as f16: each (kb,nt,lane)
//     fragment is 16 contiguous halfs (32B) -> one ds_load_b128 pair.
//   - B fragments are software-pipelined (ping-pong): the loads for step s+1
//     are issued before the WMMA of step s, so each WMMA waits with dscnt<=2
//     instead of 0 and LDS latency hides behind the matrix op.
//   - A is read as float4 (global_load_b128), converted to f16 in registers,
//     optionally scaled per-row by 1/max(rowscale[r],1) (fused mean divide).
//   - Epilogue fuses bias, residual Cin, and ReLU.
//   - Safe for A==C and Cin==C (a wave reads only the rows it writes, and
//     all loads happen before the epilogue stores).
// ---------------------------------------------------------------------------
template <int K, int LDB, int NT, bool RS>
__global__ __launch_bounds__(128)
void wmma_gemm_f16(const float* A, int lda,
                   const float* B,
                   const float* bias,
                   const float* Cin,
                   const float* rowscale,
                   float* C, int ldc,
                   int M, int do_relu)
{
    constexpr int KB = K / 32;                       // 32-wide K tiles
    __shared__ __align__(32) _Float16 Bs[KB * NT * 512];

    const int tid  = threadIdx.x;
    const int wave = tid >> 5;
    const int lane = tid & 31;
    const int tile = blockIdx.x * 4 + wave;          // 16-row tile (wave-uniform)
    const bool active = (tile * 16 < M);

    // ---- Stage + swizzle B into LDS (f32 -> f16), cooperative, coalesced reads
    for (int idx = tid; idx < K * NT * 16; idx += 128) {
        const int k  = idx / (NT * 16);
        const int n  = idx - k * (NT * 16);
        const int kb = k >> 5, rem = k & 31;
        const int hs = (rem >> 3) & 1, e = (rem & 7) + ((rem >> 4) << 3);
        const int nt = n >> 4, lr = n & 15;
        Bs[((kb * NT + nt) * 32 + hs * 16 + lr) * 16 + e] =
            (_Float16)B[(size_t)k * LDB + n];
    }
    __syncthreads();
    if (!active) return;                              // wave-uniform: EXEC stays all-ones

    const int lr = lane & 15;
    const int hs = lane >> 4;
    const int arow = tile * 16 + lr;                  // A row this lane feeds
    const float* Ap = A + (size_t)arow * lda;
    float scl = 1.0f;
    if (RS) scl = 1.0f / fmaxf(rowscale[arow], 1.0f);

    // B fragment fetch helper: 32 contiguous bytes per lane
    auto ldsfrag = [&](int kb, int nt) -> v16h {
        return *(const v16h*)(&Bs[((kb * NT + nt) * 32 + lane) * 16]);
    };

    v8f acc[NT] = {};
    v16h bcur = ldsfrag(0, 0);                        // prologue of the pipeline

    #pragma unroll
    for (int kb = 0; kb < KB; ++kb) {
        // A fragment (16x32 f16): e<8 -> k=kb*32+8*hs+e ; e>=8 -> +16
        const float* base = Ap + kb * 32 + 8 * hs;
        float4 c00 = ((const float4*)base)[0];
        float4 c01 = ((const float4*)base)[1];
        float4 c10 = ((const float4*)(base + 16))[0];
        float4 c11 = ((const float4*)(base + 16))[1];
        if (RS) {
            c00.x *= scl; c00.y *= scl; c00.z *= scl; c00.w *= scl;
            c01.x *= scl; c01.y *= scl; c01.z *= scl; c01.w *= scl;
            c10.x *= scl; c10.y *= scl; c10.z *= scl; c10.w *= scl;
            c11.x *= scl; c11.y *= scl; c11.z *= scl; c11.w *= scl;
        }
        v16h a;
        a[0]  = (_Float16)c00.x; a[1]  = (_Float16)c00.y;
        a[2]  = (_Float16)c00.z; a[3]  = (_Float16)c00.w;
        a[4]  = (_Float16)c01.x; a[5]  = (_Float16)c01.y;
        a[6]  = (_Float16)c01.z; a[7]  = (_Float16)c01.w;
        a[8]  = (_Float16)c10.x; a[9]  = (_Float16)c10.y;
        a[10] = (_Float16)c10.z; a[11] = (_Float16)c10.w;
        a[12] = (_Float16)c11.x; a[13] = (_Float16)c11.y;
        a[14] = (_Float16)c11.z; a[15] = (_Float16)c11.w;

        #pragma unroll
        for (int nt = 0; nt < NT; ++nt) {
            const bool last = (kb == KB - 1) && (nt == NT - 1);
            v16h bnxt;
            if (!last) {
                const int nkb = (nt == NT - 1) ? kb + 1 : kb;
                const int nnt = (nt == NT - 1) ? 0 : nt + 1;
                bnxt = ldsfrag(nkb, nnt);             // issue next loads first
            }
            acc[nt] = __builtin_amdgcn_wmma_f32_16x16x32_f16(
                false, a, false, bcur, (short)0, acc[nt], false, false);
            if (!last) bcur = bnxt;
        }
    }

    // ---- Epilogue. C/D layout: vgpr v -> row = tile*16 + v + 8*hs, col = nt*16 + lr
    #pragma unroll
    for (int nt = 0; nt < NT; ++nt) {
        const int n  = nt * 16 + lr;
        const float bv = bias ? bias[n] : 0.0f;
        #pragma unroll
        for (int v = 0; v < 8; ++v) {
            const int r = tile * 16 + v + 8 * hs;
            const size_t off = (size_t)r * ldc + n;
            float x = acc[nt][v] + bv;
            if (Cin) x += Cin[off];
            if (do_relu) x = fmaxf(x, 0.0f);
            C[off] = x;
        }
    }
}

// ---------------------------------------------------------------------------
// Graph mean-aggregation: one wave per edge. Each lane gathers a float4 of
// emb[src] (512B coalesced per edge) and does 4 global f32 atomic adds into
// agg[dst]; lane 0 bumps the degree counter.
// ---------------------------------------------------------------------------
__global__ __launch_bounds__(256)
void sage_scatter(const long long* ei, const float* emb,
                  float* agg, float* deg, int Ecnt)
{
    const int edge = blockIdx.x * 8 + (threadIdx.x >> 5);
    const int lane = threadIdx.x & 31;
    if (edge >= Ecnt) return;                    // wave-uniform
    const long long s = ei[edge];
    const long long d = ei[(size_t)Ecnt + edge];
    const float4 v = ((const float4*)(emb + (size_t)s * 128))[lane];
    float* dp = agg + (size_t)d * 128 + lane * 4;
    atomicAdd(dp + 0, v.x);
    atomicAdd(dp + 1, v.y);
    atomicAdd(dp + 2, v.z);
    atomicAdd(dp + 3, v.w);
    if (lane == 0) atomicAdd(deg + d, 1.0f);
}

__global__ void zero_f32(float* p, int n)
{
    int i = blockIdx.x * blockDim.x + threadIdx.x;
    if (i < n) p[i] = 0.0f;
}

// Final scalar heads: out_t{1,0}[u] = relu(h_t{1,0}[u,:] . W + b)
__global__ void out_heads(const float* h1, const float* h0,
                          const float* ot_W, const float* ot_b,
                          const float* oc_W, const float* oc_b,
                          float* o1, float* o0, int NUc)
{
    int u = blockIdx.x * blockDim.x + threadIdx.x;
    if (u >= NUc) return;
    float a = 0.0f, b = 0.0f;
    const float* p1 = h1 + (size_t)u * 64;
    const float* p0 = h0 + (size_t)u * 64;
    #pragma unroll 8
    for (int j = 0; j < 64; ++j) {
        a += p1[j] * ot_W[j];
        b += p0[j] * oc_W[j];
    }
    o1[u] = fmaxf(a + ot_b[0], 0.0f);
    o0[u] = fmaxf(b + oc_b[0], 0.0f);
}

// ---------------------------------------------------------------------------
template <int K, int LDB, int NT, bool RS>
static inline void launch_gemm(hipStream_t s, const float* A, int lda,
                               const float* B, const float* bias,
                               const float* Cin, const float* rowscale,
                               float* C, int ldc, int M, int relu)
{
    const int mtiles = M / 16;                   // all M here are multiples of 16
    hipLaunchKernelGGL((wmma_gemm_f16<K, LDB, NT, RS>),
                       dim3((mtiles + 3) / 4), dim3(128), 0, s,
                       A, lda, B, bias, Cin, rowscale, C, ldc, M, relu);
}

extern "C" void kernel_launch(void* const* d_in, const int* in_sizes, int n_in,
                              void* d_out, int out_size, void* d_ws, size_t ws_size,
                              hipStream_t stream)
{
    const float*     xu     = (const float*)d_in[0];
    const float*     xp     = (const float*)d_in[1];
    const long long* ei     = (const long long*)d_in[2];
    const float*     ue_W   = (const float*)d_in[3];
    const float*     ue_b   = (const float*)d_in[4];
    const float*     ie_W   = (const float*)d_in[5];
    const float*     ie_b   = (const float*)d_in[6];
    const float*     Wl     = (const float*)d_in[7];   // [2,128,128]
    const float*     bl     = (const float*)d_in[8];   // [2,128]
    const float*     Wr     = (const float*)d_in[9];   // [2,128,128]
    const float*     hc1_W  = (const float*)d_in[10];  // [384,128]
    const float*     hc1_b  = (const float*)d_in[11];
    const float*     hc2_W  = (const float*)d_in[12];  // [128,128]
    const float*     hc2_b  = (const float*)d_in[13];
    const float*     hctl_W = (const float*)d_in[14];  // [128,64]
    const float*     hctl_b = (const float*)d_in[15];
    const float*     htr_W  = (const float*)d_in[16];  // [128,64]
    const float*     htr_b  = (const float*)d_in[17];
    const float*     oc_W   = (const float*)d_in[18];  // [64]
    const float*     oc_b   = (const float*)d_in[19];
    const float*     ot_W   = (const float*)d_in[20];  // [64]
    const float*     ot_b   = (const float*)d_in[21];

    const int NU = in_sizes[0] / 128;
    const int NP = in_sizes[1] / 64;
    const int E  = in_sizes[2] / 2;
    const int N  = NU + NP;
    const int nh = N * 128;                      // floats per [N,H] buffer

    // Workspace layout (floats):
    float* ws   = (float*)d_ws;
    float* emb0 = ws;                            // [N,128]  rows 0..NU-1 = xu_e
    float* emb1 = ws + (size_t)nh;               // [N,128]  after SAGE layer 1
    float* emb2 = ws + 2 * (size_t)nh;           // [N,128]  after SAGE layer 2
    float* agg  = ws + 3 * (size_t)nh;           // [N,128]  scatter accum / lin_l temp / hidden1
    float* deg  = ws + 4 * (size_t)nh;           // [N]      degree counts (adjacent to agg)
    float* hid1 = agg;                           // reuse (agg dead after layer 2)
    float* hid2 = emb1;                          // reuse (emb1 dead after hc1)

    float* out  = (float*)d_out;                 // (out_t1, out_t0, h_t1, h_t0) flat
    float* o_t1 = out;
    float* o_t0 = out + (size_t)NU;
    float* h_t1 = out + 2 * (size_t)NU;
    float* h_t0 = out + 2 * (size_t)NU + 64 * (size_t)NU;

    // 1) Node embeddings (no relu)
    launch_gemm<128, 128, 8, false>(stream, xu, 128, ue_W, ue_b, nullptr, nullptr,
                                    emb0, 128, NU, 0);
    launch_gemm<64, 128, 8, false>(stream, xp, 64, ie_W, ie_b, nullptr, nullptr,
                                   emb0 + (size_t)NU * 128, 128, NP, 0);

    // 2) Two SAGE layers: emb_out = relu( (sum/deg)@Wl + bl + emb_in@Wr )
    for (int layer = 0; layer < 2; ++layer) {
        const float* embin  = (layer == 0) ? emb0 : emb1;
        float*       embout = (layer == 0) ? emb1 : emb2;
        const int zn = nh + N;                   // agg and deg are contiguous
        hipLaunchKernelGGL(zero_f32, dim3((zn + 255) / 256), dim3(256), 0, stream, agg, zn);
        hipLaunchKernelGGL(sage_scatter, dim3((E + 7) / 8), dim3(256), 0, stream,
                           ei, embin, agg, deg, E);
        // agg = (agg / max(deg,1)) @ Wl[layer] + bl[layer]   (fused mean, in-place)
        launch_gemm<128, 128, 8, true>(stream, agg, 128, Wl + (size_t)layer * 16384,
                                       bl + (size_t)layer * 128, nullptr, deg,
                                       agg, 128, N, 0);
        // embout = relu( embin @ Wr[layer] + agg )
        launch_gemm<128, 128, 8, false>(stream, embin, 128, Wr + (size_t)layer * 16384,
                                        nullptr, agg, nullptr, embout, 128, N, 1);
    }

    // 3) hidden1 = relu( [xu_e | l1_u | l2_u] @ hc1_W + hc1_b )  as 3 chained GEMMs
    launch_gemm<128, 128, 8, false>(stream, emb0, 128, hc1_W,         hc1_b,  nullptr, nullptr,
                                    hid1, 128, NU, 0);
    launch_gemm<128, 128, 8, false>(stream, emb1, 128, hc1_W + 16384, nullptr, hid1, nullptr,
                                    hid1, 128, NU, 0);
    launch_gemm<128, 128, 8, false>(stream, emb2, 128, hc1_W + 32768, nullptr, hid1, nullptr,
                                    hid1, 128, NU, 1);

    // 4) hidden2 = relu( hidden1 @ hc2_W + hc2_b )
    launch_gemm<128, 128, 8, false>(stream, hid1, 128, hc2_W, hc2_b, nullptr, nullptr,
                                    hid2, 128, NU, 1);

    // 5) Branches straight into d_out: h_t1 / h_t0 = relu( hidden2 @ W + b ), Ncols=64
    launch_gemm<128, 64, 4, false>(stream, hid2, 128, htr_W,  htr_b,  nullptr, nullptr,
                                   h_t1, 64, NU, 1);
    launch_gemm<128, 64, 4, false>(stream, hid2, 128, hctl_W, hctl_b, nullptr, nullptr,
                                   h_t0, 64, NU, 1);

    // 6) Scalar heads
    hipLaunchKernelGGL(out_heads, dim3((NU + 255) / 256), dim3(256), 0, stream,
                       h_t1, h_t0, ot_W, ot_b, oc_W, oc_b, o_t1, o_t0, NU);
}
